// torchGRU_cell_my_implementation_24627342475575
// MI455X (gfx1250) — matrix operand order repaired and checked by reading, not compiled, not run on hardware
//
#include <hip/hip_runtime.h>
#include <hip/hip_bf16.h>
#include <math.h>

// ---------------------------------------------------------------------------
// Problem constants (from the reference)
// ---------------------------------------------------------------------------
#define NT     200
#define NGRID  1024
#define NX     256
#define HID    512
#define KDIM   (NX + HID)          // 768  : A = [x_t | h]
#define NCOLS  (2*HID + HID + HID) // 2048 : [z | r | nx | nh]

typedef __attribute__((ext_vector_type(16))) _Float16 v16h;
typedef __attribute__((ext_vector_type(8)))  _Float16 h8;
typedef __attribute__((ext_vector_type(8)))  float    v8f;

// ---------------------------------------------------------------------------
// Weight pre-pack: Bpack[cell] is [KDIM x NCOLS] f16 stored in 32-K panels:
//   idx = ((k/32)*NCOLS + n)*32 + (k%32)
// so a WMMA B-fragment lane (fixed n, 16 consecutive k) is one contiguous
// 32-byte load.  Block structure:
//   n in [0,1024)      : gates  -> k<256: W_ih[n][k]        else W_hh[n][k-256]
//   n in [1024,1536)   : nx     -> k<256: Wn_ih[n-1024][k]  else 0
//   n in [1536,2048)   : nh     -> k<256: 0                 else Wn_hh[n-1536][k-256]
// ---------------------------------------------------------------------------
__global__ void gru_prepack_B_kernel(const float* __restrict__ W_ih,
                                     const float* __restrict__ W_hh,
                                     const float* __restrict__ Wn_ih,
                                     const float* __restrict__ Wn_hh,
                                     _Float16* __restrict__ Bp) {
  int tid = blockIdx.x * blockDim.x + threadIdx.x;
  const int per_cell = KDIM * NCOLS;
  if (tid >= 2 * per_cell) return;
  int cell = tid / per_cell;
  int rem  = tid - cell * per_cell;
  int k = rem / NCOLS;
  int n = rem - k * NCOLS;

  float v = 0.0f;
  if (n < 2 * HID) {                       // gate columns (z|r)
    v = (k < NX) ? W_ih[(size_t)cell * 2 * HID * NX + (size_t)n * NX + k]
                 : W_hh[(size_t)cell * 2 * HID * HID + (size_t)n * HID + (k - NX)];
  } else if (n < 3 * HID) {                // nx columns (x only)
    int j = n - 2 * HID;
    v = (k < NX) ? Wn_ih[(size_t)cell * HID * NX + (size_t)j * NX + k] : 0.0f;
  } else {                                 // nh columns (h only)
    int j = n - 3 * HID;
    v = (k < NX) ? 0.0f
                 : Wn_hh[(size_t)cell * HID * HID + (size_t)j * HID + (k - NX)];
  }
  size_t idx = (size_t)cell * per_cell +
               (((size_t)(k >> 5) * NCOLS + n) << 5) + (k & 31);
  Bp[idx] = (_Float16)v;
}

__global__ void gru_prepack_bias_kernel(const float* __restrict__ b_ih,
                                        const float* __restrict__ b_hh,
                                        const float* __restrict__ bn_ih,
                                        const float* __restrict__ bn_hh,
                                        float* __restrict__ biasP) {
  int tid = blockIdx.x * blockDim.x + threadIdx.x;
  if (tid >= 2 * NCOLS) return;
  int cell = tid / NCOLS;
  int n    = tid - cell * NCOLS;
  float v;
  if (n < 2 * HID)      v = b_ih[cell * 2 * HID + n] + b_hh[cell * 2 * HID + n];
  else if (n < 3 * HID) v = bn_ih[cell * HID + (n - 2 * HID)];
  else                  v = bn_hh[cell * HID + (n - 3 * HID)];
  biasP[tid] = v;
}

// Zero h state and the h-columns of the f16 A buffer.
__global__ void gru_init_kernel(float* __restrict__ h32,
                                _Float16* __restrict__ A16) {
  int tid = blockIdx.x * blockDim.x + threadIdx.x;
  if (tid >= NGRID * HID) return;
  int g = tid >> 9;          // /512
  int j = tid & (HID - 1);
  h32[tid] = 0.0f;
  A16[(size_t)g * KDIM + NX + j] = (_Float16)0.0f;
}

// Per-step: write f16(x[t]) into A columns [0,256).
__global__ void gru_xpack_kernel(const float* __restrict__ x, int t,
                                 _Float16* __restrict__ A16) {
  int tid = blockIdx.x * blockDim.x + threadIdx.x;
  if (tid >= NGRID * NX) return;
  int g = tid >> 8;          // /256
  int c = tid & (NX - 1);
  A16[(size_t)g * KDIM + c] =
      (_Float16)x[((size_t)t * NGRID + g) * NX + c];
}

// ---------------------------------------------------------------------------
// WMMA GEMM:  C32[NGRID x NCOLS] = A16[NGRID x KDIM] @ Bpack + bias
// Block: 256 threads = 8 waves, tile 128(M) x 128(N).
// Wave tile: 64(M) x 32(N) = 4x2 fragments of v_wmma_f32_16x16x32_f16.
// A-fragment lane layout (this code's contiguous k order, same as B; not the ISA table order):
//   m = lane%16 ; kbase = (lane/16)*16 ; halves 0..7 -> k=kbase+0..7,
//   halves 8..15 -> k=kbase+8..15   => two contiguous 16B row loads.
// B-fragment lane layout: n = lane%16 ; k = (lane/16)*16 + j (j=0..15)
//   => one contiguous 32B load from the 32-K panel layout.
// ---------------------------------------------------------------------------
__global__ void __launch_bounds__(256)
gru_wmma_gemm_kernel(const _Float16* __restrict__ A16,
                     const _Float16* __restrict__ Bp,
                     const float* __restrict__ bias,
                     float* __restrict__ C32) {
  const int lane = threadIdx.x & 31;
  const int wid  = threadIdx.x >> 5;
  const int wy   = wid & 1;   // M half within block (2 x 64)
  const int wx   = wid >> 1;  // N quarter within block (4 x 32)

  const int Mb = blockIdx.y * 128 + wy * 64;   // wave M base
  const int Nb = blockIdx.x * 128 + wx * 32;   // wave N base

  const int lm = lane & 15;        // A row / B col / C col sub-index
  const int lh = lane >> 4;        // lane half

  v8f acc[4][2];
#pragma unroll
  for (int mi = 0; mi < 4; ++mi)
#pragma unroll
    for (int ni = 0; ni < 2; ++ni) acc[mi][ni] = (v8f){0,0,0,0,0,0,0,0};

  for (int kk = 0; kk < KDIM / 32; ++kk) {
    const int Kb = kk * 32;

    // --- B fragments: contiguous 16 halves per lane ---
    v16h bfrag[2];
#pragma unroll
    for (int ni = 0; ni < 2; ++ni) {
      const int n = Nb + ni * 16 + lm;
      const size_t base = (((size_t)kk * NCOLS + n) << 5) + (size_t)lh * 16;
      bfrag[ni] = *(const v16h*)(Bp + base);
    }

    // --- A fragments: two 16B loads per lane ---
    v16h afrag[4];
#pragma unroll
    for (int mi = 0; mi < 4; ++mi) {
      const int m  = Mb + mi * 16 + lm;
      const int kb = Kb + lh * 16;
      const h8 lo = *(const h8*)(A16 + (size_t)m * KDIM + kb);
      const h8 hi = *(const h8*)(A16 + (size_t)m * KDIM + kb + 8);
      afrag[mi] = __builtin_shufflevector(lo, hi, 0, 1, 2, 3, 4, 5, 6, 7,
                                          8, 9, 10, 11, 12, 13, 14, 15);
    }

#pragma unroll
    for (int mi = 0; mi < 4; ++mi)
#pragma unroll
      for (int ni = 0; ni < 2; ++ni)
        acc[mi][ni] = __builtin_amdgcn_wmma_f32_16x16x32_f16(
            false, afrag[mi], false, bfrag[ni], (short)0, acc[mi][ni],
            false, false);
  }

  // Epilogue: D layout (ISA 7.12.2): col = Nb+ni*16+lm, row = v + lh*8.
#pragma unroll
  for (int mi = 0; mi < 4; ++mi)
#pragma unroll
    for (int ni = 0; ni < 2; ++ni) {
      const int col = Nb + ni * 16 + lm;
      const float b = bias[col];
#pragma unroll
      for (int v = 0; v < 8; ++v) {
        const int row = Mb + mi * 16 + lh * 8 + v;
        C32[(size_t)row * NCOLS + col] = acc[mi][ni][v] + b;
      }
    }
}

// ---------------------------------------------------------------------------
// GRU elementwise update (per cell).  NOTE: reference applies NO tanh to n.
//   z = sig(C[:,j]) ; r = sig(C[:,512+j])
//   n = C[:,1024+j] + r*C[:,1536+j]
//   h = z*h + (1-z)*n   -> writes fp32 state AND f16 A h-columns.
// ---------------------------------------------------------------------------
__global__ void gru_update_kernel(const float* __restrict__ C32,
                                  float* __restrict__ h32,
                                  _Float16* __restrict__ A16) {
  int tid = blockIdx.x * blockDim.x + threadIdx.x;
  if (tid >= NGRID * HID) return;
  int g = tid >> 9;
  int j = tid & (HID - 1);
  const float* row = C32 + (size_t)g * NCOLS;
  float z  = 1.0f / (1.0f + expf(-row[j]));
  float r  = 1.0f / (1.0f + expf(-row[HID + j]));
  float n  = row[2 * HID + j] + r * row[3 * HID + j];
  float hv = h32[tid];
  float hn = z * hv + (1.0f - z) * n;
  h32[tid] = hn;
  A16[(size_t)g * KDIM + NX + j] = (_Float16)hn;
}

// ---------------------------------------------------------------------------
// Output projection for one timestep: out[t,g] = dot(h[g,:], W_out) + b_out.
// One wave per grid row; wave-shuffle reduction.
// ---------------------------------------------------------------------------
__global__ void gru_outproj_kernel(const float* __restrict__ h32,
                                   const float* __restrict__ W_out,
                                   const float* __restrict__ b_out,
                                   float* __restrict__ out, int t) {
  const int lane = threadIdx.x & 31;
  const int wid  = threadIdx.x >> 5;
  const int g    = blockIdx.x * 8 + wid;   // 8 waves per 256-thread block
  float s = 0.0f;
#pragma unroll
  for (int j = lane; j < HID; j += 32)
    s += h32[(size_t)g * HID + j] * W_out[j];
#pragma unroll
  for (int off = 16; off > 0; off >>= 1)
    s += __shfl_down(s, off, 32);
  if (lane == 0)
    out[(size_t)t * NGRID + g] = s + b_out[0];
}

// ---------------------------------------------------------------------------
// Host launcher.  Workspace layout (bytes):
//   [0)                Bpack  : 2 * 768*2048 f16   = 6,291,456
//   [6,291,456)        biasP  : 2 * 2048 f32       =    16,384
//   [6,307,840)        A16    : 1024*768 f16       = 1,572,864
//   [7,880,704)        C32    : 1024*2048 f32      = 8,388,608
//   [16,269,312)       h32    : 1024*512 f32       = 2,097,152
//   total ~ 18.4 MB
// ---------------------------------------------------------------------------
extern "C" void kernel_launch(void* const* d_in, const int* in_sizes, int n_in,
                              void* d_out, int out_size, void* d_ws, size_t ws_size,
                              hipStream_t stream) {
  const float* x     = (const float*)d_in[0];
  const float* W_ih  = (const float*)d_in[1];
  const float* b_ih  = (const float*)d_in[2];
  const float* W_hh  = (const float*)d_in[3];
  const float* b_hh  = (const float*)d_in[4];
  const float* Wn_ih = (const float*)d_in[5];
  const float* bn_ih = (const float*)d_in[6];
  const float* Wn_hh = (const float*)d_in[7];
  const float* bn_hh = (const float*)d_in[8];
  const float* W_out = (const float*)d_in[9];
  const float* b_out = (const float*)d_in[10];
  float* out = (float*)d_out;

  char* ws = (char*)d_ws;
  _Float16* Bp    = (_Float16*)(ws);
  float*    biasP = (float*)(ws + 6291456);
  _Float16* A16   = (_Float16*)(ws + 6307840);
  float*    C32   = (float*)(ws + 7880704);
  float*    h32   = (float*)(ws + 16269312);

  const int per_cell = KDIM * NCOLS;

  // One-time (per call) weight packing + state init.
  gru_prepack_B_kernel<<<(2 * per_cell + 255) / 256, 256, 0, stream>>>(
      W_ih, W_hh, Wn_ih, Wn_hh, Bp);
  gru_prepack_bias_kernel<<<(2 * NCOLS + 255) / 256, 256, 0, stream>>>(
      b_ih, b_hh, bn_ih, bn_hh, biasP);
  gru_init_kernel<<<(NGRID * HID + 255) / 256, 256, 0, stream>>>(h32, A16);

  const dim3 gemm_grid(NCOLS / 128, NGRID / 128);  // 16 x 8 blocks

  for (int t = 0; t < NT; ++t) {
    gru_xpack_kernel<<<(NGRID * NX + 255) / 256, 256, 0, stream>>>(x, t, A16);
    for (int cell = 0; cell < 2; ++cell) {
      gru_wmma_gemm_kernel<<<gemm_grid, 256, 0, stream>>>(
          A16, Bp + (size_t)cell * per_cell, biasP + cell * NCOLS, C32);
      gru_update_kernel<<<(NGRID * HID + 255) / 256, 256, 0, stream>>>(
          C32, h32, A16);
    }
    gru_outproj_kernel<<<NGRID / 8, 256, 0, stream>>>(h32, W_out, b_out, out, t);
  }
}
